// MixedCausalAttention_50826642980980
// MI455X (gfx1250) — matrix operand order repaired and checked by reading, not compiled, not run on hardware
//
#include <hip/hip_runtime.h>
#include <hip/hip_bf16.h>
#include <math.h>

typedef __bf16 bf16;
typedef __attribute__((ext_vector_type(16))) __bf16 v16bf;
typedef __attribute__((ext_vector_type(8)))  __bf16 v8bf;
typedef __attribute__((ext_vector_type(8)))  float  v8f;

#define D_MODEL  1024
#define NUM_HEADS 16
#define HEAD_DIM  64
#define SEQ_TC   2032
#define NS_TC    16
#define SEQ_LEN  2048
#define BATCH    4

static __device__ __forceinline__ v16bf cat16(v8bf lo, v8bf hi) {
  return __builtin_shufflevector(lo, hi, 0,1,2,3,4,5,6,7,8,9,10,11,12,13,14,15);
}

// Async DMA: copy 16B from global straight into LDS (no VGPR staging).
// LDS address = low 32 bits of the generic pointer.
static __device__ __forceinline__ void async_cp16(void* lds_dst, const void* gsrc) {
  unsigned lds = (unsigned)(unsigned long long)lds_dst;
  unsigned long long ga = (unsigned long long)gsrc;
  asm volatile("global_load_async_to_lds_b128 %0, %1, off"
               :: "v"(lds), "v"(ga) : "memory");
}
static __device__ __forceinline__ void wait_async0() {
  asm volatile("s_wait_asynccnt 0" ::: "memory");
}

// ---------------------------------------------------------------------------
// Kernel 1: x + positional encoding -> bf16
// ---------------------------------------------------------------------------
__global__ __launch_bounds__(256) void pe_add_bf16(const float* __restrict__ x,
                                                   bf16* __restrict__ xb, int total) {
  int i = blockIdx.x * blockDim.x + threadIdx.x;
  if (i >= total) return;
  int d = i & (D_MODEL - 1);
  int s = (i >> 10) & (SEQ_LEN - 1);
  float div = __expf(-0.0089944732f * (float)(d & ~1));   // exp(-d*ln(1e4)/D)
  float ang = (float)s * div;
  float pe  = (d & 1) ? __cosf(ang) : __sinf(ang);
  xb[i] = (bf16)(x[i] + pe);
}

// ---------------------------------------------------------------------------
// Kernel 2: generic C = A(MxK,bf16) * W(NxK,f32->bf16)^T + bias, WMMA bf16
// Output element (m, o=h*64+d) stored at: out_base + m*o_rs + h*o_hs + d*o_ds
// ---------------------------------------------------------------------------
#define BM 128
#define BN 128
#define BK 32
#define ASTR 40   // padded LDS row stride (halves): 80B, bank-conflict-free

__global__ __launch_bounds__(256) void gemm_xwT_wmma(
    const bf16* __restrict__ A, long long a_rs, int M,
    const float* __restrict__ W, const float* __restrict__ bias,
    void* __restrict__ out, long long out_base, long long o_rs,
    long long o_hs, long long o_ds, int out_f32)
{
  __shared__ bf16 As[2][BM][ASTR];
  __shared__ bf16 Bs[2][BN][ASTR];

  const int tid  = threadIdx.x;
  const int lane = tid & 31;
  const int wid  = tid >> 5;
  const int wm   = wid >> 2;        // 0..1
  const int wn   = wid & 3;         // 0..3
  const int l15  = lane & 15;
  const int half = lane >> 4;       // 0 or 1
  const int row0 = blockIdx.x * BM;
  const int n0   = blockIdx.y * BN;

  v8f acc[4][2];
  const v8f vzero = {0.f,0.f,0.f,0.f,0.f,0.f,0.f,0.f};
  for (int mf = 0; mf < 4; ++mf)
    for (int nf = 0; nf < 2; ++nf) acc[mf][nf] = vzero;

  auto stage = [&](int kt, int buf) {
    const int k0 = kt * BK;
    const int r  = tid >> 1;
    const int c  = tid & 1;
    // A tile: async DMA global->LDS. Rows past M are clamped to row M-1:
    // they only feed C rows >= M, which are never stored.
    int rr = row0 + r; if (rr >= M) rr = M - 1;
    const bf16* src = A + (long long)rr * a_rs + k0 + c * 16;
    async_cp16(&As[buf][r][c * 16],     src);
    async_cp16(&As[buf][r][c * 16 + 8], src + 8);
    // W tile: fp32 -> bf16 while staging (VALU path)
    const float* ws = W + (long long)(n0 + r) * D_MODEL + k0 + c * 16;
    if (k0 + BK < D_MODEL) __builtin_prefetch(ws + BK, 0, 1);
    v8bf b0, b1;
    for (int i = 0; i < 8; ++i) { b0[i] = (bf16)ws[i]; b1[i] = (bf16)ws[i + 8]; }
    *(v8bf*)&Bs[buf][r][c * 16]     = b0;
    *(v8bf*)&Bs[buf][r][c * 16 + 8] = b1;
  };

  stage(0, 0);
  const int nk = D_MODEL / BK;
  for (int kt = 0; kt < nk; ++kt) {
    const int buf = kt & 1;
    wait_async0();          // async fills of buffer `buf` complete (ASYNCcnt)
    __syncthreads();        // compiler adds s_wait_dscnt for B-tile ds_stores
    if (kt + 1 < nk) stage(kt + 1, buf ^ 1);

    v16bf af[4], bfv[2];
    for (int mf = 0; mf < 4; ++mf) {
      const bf16* p = &As[buf][wm * 64 + mf * 16 + l15][half * 8];
      af[mf] = cat16(*(const v8bf*)p, *(const v8bf*)(p + 16));
    }
    for (int nf = 0; nf < 2; ++nf) {
      const bf16* p = &Bs[buf][wn * 32 + nf * 16 + l15][half * 8];
      bfv[nf] = cat16(*(const v8bf*)p, *(const v8bf*)(p + 16));
    }
    for (int mf = 0; mf < 4; ++mf)
      for (int nf = 0; nf < 2; ++nf)
        acc[mf][nf] = __builtin_amdgcn_wmma_f32_16x16x32_bf16(
            false, af[mf], false, bfv[nf], (short)0, acc[mf][nf], false, false);
  }

  // epilogue: bias + store via generic address formula
  for (int mf = 0; mf < 4; ++mf) {
    for (int nf = 0; nf < 2; ++nf) {
      const int o = n0 + wn * 32 + nf * 16 + l15;
      const float bv = bias[o];
      const long long obase = out_base + (long long)(o >> 6) * o_hs
                                       + (long long)(o & 63) * o_ds;
      for (int r = 0; r < 8; ++r) {
        const int m = row0 + wm * 64 + mf * 16 + r + half * 8;
        if (m < M) {
          const float v = acc[mf][nf][r] + bv;
          const long long idx = obase + (long long)m * o_rs;
          if (out_f32) ((float*)out)[idx] = v;
          else         ((bf16*)out)[idx]  = (bf16)v;
        }
      }
    }
  }
}

// ---------------------------------------------------------------------------
// Kernel 3: flash attention (causal), bf16 WMMA, f32 online softmax
// q,k: [b][h][s][hd] bf16 ; vt: [b][h][hd][s] bf16 ; ctx: [b][s][h*64+d] bf16
// block = 128 threads (4 waves); each block: 64 query rows of one (b,h)
// ---------------------------------------------------------------------------
#define PSTR 88   // padded LDS row stride (halves): 176B, conflict-free b128

__global__ __launch_bounds__(128) void flash_attn_wmma(
    const bf16* __restrict__ q, const bf16* __restrict__ k,
    const bf16* __restrict__ vt, bf16* __restrict__ ctx)
{
  __shared__ bf16 Ks[64][PSTR];      // [key_local][d]
  __shared__ bf16 Vs[64][PSTR];      // [d][key_local]
  __shared__ bf16 Ps[4][16][PSTR];   // per-wave P tile [m][key_local]

  const int tid  = threadIdx.x;
  const int lane = tid & 31;
  const int w    = tid >> 5;
  const int l15  = lane & 15;
  const int half = lane >> 4;
  const int qt = blockIdx.x, h = blockIdx.y, b = blockIdx.z;
  const int q0 = qt * 64;

  const bf16* qb = q  + (long long)(b * NUM_HEADS + h) * SEQ_LEN * HEAD_DIM;
  const bf16* kb = k  + (long long)(b * NUM_HEADS + h) * SEQ_LEN * HEAD_DIM;
  const bf16* vb = vt + (long long)(b * NUM_HEADS + h) * HEAD_DIM * SEQ_LEN;

  // Q A-fragments for this wave's 16 rows (kept in registers)
  v16bf aq[2];
  {
    const bf16* qp = qb + (long long)(q0 + w * 16 + l15) * HEAD_DIM + half * 8;
    aq[0] = cat16(*(const v8bf*)qp,        *(const v8bf*)(qp + 16));
    aq[1] = cat16(*(const v8bf*)(qp + 32), *(const v8bf*)(qp + 48));
  }

  v8f oacc[4];
  const v8f vzero = {0.f,0.f,0.f,0.f,0.f,0.f,0.f,0.f};
  for (int db = 0; db < 4; ++db) oacc[db] = vzero;
  float mst[8], lst[8];
  for (int r = 0; r < 8; ++r) { mst[r] = -1e30f; lst[r] = 0.f; }

  for (int j = 0; j <= qt; ++j) {
    { // async DMA stage of K tile and V^T tile straight into LDS
      const int r = tid >> 1, c = tid & 1;
      const bf16* ks = kb + (long long)(j * 64 + r) * HEAD_DIM + c * 32;
      async_cp16(&Ks[r][c * 32 + 0],  ks + 0);
      async_cp16(&Ks[r][c * 32 + 8],  ks + 8);
      async_cp16(&Ks[r][c * 32 + 16], ks + 16);
      async_cp16(&Ks[r][c * 32 + 24], ks + 24);
      const bf16* vs = vb + (long long)r * SEQ_LEN + j * 64 + c * 32;
      async_cp16(&Vs[r][c * 32 + 0],  vs + 0);
      async_cp16(&Vs[r][c * 32 + 8],  vs + 8);
      async_cp16(&Vs[r][c * 32 + 16], vs + 16);
      async_cp16(&Vs[r][c * 32 + 24], vs + 24);
    }
    wait_async0();
    __syncthreads();

    // S = Q * K^T  (16 x 64 per wave)
    v8f sf[4];
    for (int nb = 0; nb < 4; ++nb) sf[nb] = vzero;
    for (int nb = 0; nb < 4; ++nb) {
      const int kl = nb * 16 + l15;
      for (int kk = 0; kk < 2; ++kk) {
        const bf16* p = &Ks[kl][kk * 32 + half * 8];
        v16bf bk = cat16(*(const v8bf*)p, *(const v8bf*)(p + 16));
        sf[nb] = __builtin_amdgcn_wmma_f32_16x16x32_bf16(
            false, aq[kk], false, bk, (short)0, sf[nb], false, false);
      }
    }

    // scale + causal mask + row max
    float rowm[8];
    for (int r = 0; r < 8; ++r) rowm[r] = -1e30f;
    for (int nb = 0; nb < 4; ++nb) {
      const int kidx = j * 64 + nb * 16 + l15;
      for (int r = 0; r < 8; ++r) {
        const int qidx = q0 + w * 16 + r + half * 8;
        float s = sf[nb][r] * 0.125f;           // 1/sqrt(64)
        s = (kidx <= qidx) ? s : -1e30f;
        sf[nb][r] = s;
        rowm[r] = fmaxf(rowm[r], s);
      }
    }
    for (int r = 0; r < 8; ++r)
      for (int off = 8; off > 0; off >>= 1)
        rowm[r] = fmaxf(rowm[r], __shfl_xor(rowm[r], off, 32));

    float rsum[8];
    for (int r = 0; r < 8; ++r) {
      const float mn = fmaxf(mst[r], rowm[r]);
      const float al = __expf(mst[r] - mn);
      mst[r] = mn;
      lst[r] *= al;
      rsum[r] = 0.f;
      for (int db = 0; db < 4; ++db) oacc[db][r] *= al;
    }
    // P = exp(S - m), write bf16 tile to LDS (transpose via LDS for A-frags)
    for (int nb = 0; nb < 4; ++nb)
      for (int r = 0; r < 8; ++r) {
        const float p = __expf(sf[nb][r] - mst[r]);
        rsum[r] += p;
        Ps[w][r + half * 8][nb * 16 + l15] = (bf16)p;
      }
    for (int r = 0; r < 8; ++r) {
      for (int off = 8; off > 0; off >>= 1)
        rsum[r] += __shfl_xor(rsum[r], off, 32);
      lst[r] += rsum[r];
    }

    // O += P * V   (B operand from V^T tile: contiguous in key)
    v16bf ap[2];
    {
      const bf16* pp = &Ps[w][l15][half * 8];
      ap[0] = cat16(*(const v8bf*)pp,        *(const v8bf*)(pp + 16));
      ap[1] = cat16(*(const v8bf*)(pp + 32), *(const v8bf*)(pp + 48));
    }
    for (int db = 0; db < 4; ++db) {
      const int dr = db * 16 + l15;
      for (int kk = 0; kk < 2; ++kk) {
        const bf16* vp = &Vs[dr][kk * 32 + half * 8];
        v16bf bv = cat16(*(const v8bf*)vp, *(const v8bf*)(vp + 16));
        oacc[db] = __builtin_amdgcn_wmma_f32_16x16x32_bf16(
            false, ap[kk], false, bv, (short)0, oacc[db], false, false);
      }
    }
    __syncthreads();
  }

  // normalize and write context (bf16, [b][s][h*64+d])
  for (int db = 0; db < 4; ++db)
    for (int r = 0; r < 8; ++r) {
      const int srow = q0 + w * 16 + r + half * 8;
      const int col  = h * 64 + db * 16 + l15;
      ctx[((long long)b * SEQ_LEN + srow) * D_MODEL + col] =
          (bf16)(oacc[db][r] * (1.0f / lst[r]));
    }
}

// ---------------------------------------------------------------------------
// Host launcher
// ---------------------------------------------------------------------------
extern "C" void kernel_launch(void* const* d_in, const int* in_sizes, int n_in,
                              void* d_out, int out_size, void* d_ws, size_t ws_size,
                              hipStream_t stream)
{
  (void)in_sizes; (void)n_in; (void)out_size; (void)ws_size;
  const float* x   = (const float*)d_in[0];
  const float* Wq  = (const float*)d_in[1];
  const float* bq  = (const float*)d_in[2];
  const float* Wk  = (const float*)d_in[3];
  const float* bk  = (const float*)d_in[4];
  const float* Wv  = (const float*)d_in[5];
  const float* bv  = (const float*)d_in[6];
  const float* Wqn = (const float*)d_in[7];
  const float* bqn = (const float*)d_in[8];
  const float* Wkn = (const float*)d_in[9];
  const float* bkn = (const float*)d_in[10];
  const float* Wvn = (const float*)d_in[11];
  const float* bvn = (const float*)d_in[12];
  const float* Wo  = (const float*)d_in[13];
  const float* bo  = (const float*)d_in[14];

  const long long NE = (long long)BATCH * SEQ_LEN * D_MODEL;   // 8M elements
  bf16* xb  = (bf16*)d_ws;
  bf16* qw  = xb  + NE;
  bf16* kw  = qw  + NE;
  bf16* vtw = kw  + NE;
  bf16* ctx = vtw + NE;

  // 1) positional encoding + bf16 convert
  pe_add_bf16<<<(int)((NE + 255) / 256), 256, 0, stream>>>(x, xb, (int)NE);

  const long long HS = (long long)SEQ_LEN * HEAD_DIM;          // per-head plane
  const long long DD = (long long)D_MODEL * D_MODEL;

  // 2) seq-token QKV projections (shared weights), per batch
  for (int b = 0; b < BATCH; ++b) {
    const bf16* Ab = xb + (long long)b * SEQ_LEN * D_MODEL;
    const long long ob = (long long)b * SEQ_LEN * D_MODEL;     // = b*H*S*64
    dim3 g((SEQ_TC + BM - 1) / BM, D_MODEL / BN, 1);
    gemm_xwT_wmma<<<g, 256, 0, stream>>>(Ab, D_MODEL, SEQ_TC, Wq, bq, qw,
        ob, 64, HS, 1, 0);
    gemm_xwT_wmma<<<g, 256, 0, stream>>>(Ab, D_MODEL, SEQ_TC, Wk, bk, kw,
        ob, 64, HS, 1, 0);
    gemm_xwT_wmma<<<g, 256, 0, stream>>>(Ab, D_MODEL, SEQ_TC, Wv, bv, vtw,
        ob, 1, (long long)HEAD_DIM * SEQ_LEN, SEQ_LEN, 0);     // V stored transposed
  }

  // 3) ns-token QKV projections: per-position weights, M = BATCH rows
  for (int n = 0; n < NS_TC; ++n) {
    const bf16* An = xb + (long long)(SEQ_TC + n) * D_MODEL;
    const long long a_rs = (long long)SEQ_LEN * D_MODEL;       // batch stride
    dim3 g(1, D_MODEL / BN, 1);
    gemm_xwT_wmma<<<g, 256, 0, stream>>>(An, a_rs, BATCH,
        Wqn + (long long)n * DD, bqn + (long long)n * D_MODEL, qw,
        (long long)(SEQ_TC + n) * HEAD_DIM, a_rs, HS, 1, 0);
    gemm_xwT_wmma<<<g, 256, 0, stream>>>(An, a_rs, BATCH,
        Wkn + (long long)n * DD, bkn + (long long)n * D_MODEL, kw,
        (long long)(SEQ_TC + n) * HEAD_DIM, a_rs, HS, 1, 0);
    gemm_xwT_wmma<<<g, 256, 0, stream>>>(An, a_rs, BATCH,
        Wvn + (long long)n * DD, bvn + (long long)n * D_MODEL, vtw,
        (long long)(SEQ_TC + n), a_rs, (long long)HEAD_DIM * SEQ_LEN, SEQ_LEN, 0);
  }

  // 4) causal flash attention
  dim3 ga(SEQ_LEN / 64, NUM_HEADS, BATCH);
  flash_attn_wmma<<<ga, 128, 0, stream>>>(qw, kw, vtw, ctx);

  // 5) output projection -> fp32 d_out [B][S][D]
  dim3 go((BATCH * SEQ_LEN) / BM, D_MODEL / BN, 1);
  gemm_xwT_wmma<<<go, 256, 0, stream>>>(ctx, D_MODEL, BATCH * SEQ_LEN, Wo, bo,
      d_out, 0, D_MODEL, 64, 1, 1);
}
